// MultiHeadAttention_12575664243470
// MI455X (gfx1250) — compile-verified, matrix operand order checked by reading, not run on hardware
//
#include <hip/hip_runtime.h>

// ---------------------------------------------------------------------------
// MI455X (gfx1250) multi-head attention with RoPE.
// bf16 data path, f32 accumulation via v_wmma_f32_16x16x32_bf16 (wave32).
// GEMMs stage A/B tiles through LDS with the Tensor Data Mover (TDM),
// double-buffered (K-step 64), synchronized with s_wait_tensorcnt + barriers.
// sched_group_barrier pins a loads-then-WMMA pipeline (partial dscnt waits);
// __launch_bounds__(256,4) sets a 256-VGPR budget so the pipeline's live
// fragments don't spill.
// ---------------------------------------------------------------------------

typedef __bf16 bf16;
typedef __attribute__((ext_vector_type(8)))  bf16  v8bf;
typedef __attribute__((ext_vector_type(16))) bf16  v16bf;
typedef __attribute__((ext_vector_type(8)))  float v8f;
typedef __attribute__((ext_vector_type(4)))  unsigned int u32x4;
typedef __attribute__((ext_vector_type(4)))  int i32x4;
typedef __attribute__((ext_vector_type(8)))  int i32x8;

#define T_SEQ 4096
#define D_MODEL 2048
#define N_HEADS 16
#define D_HEAD 128

#if defined(__has_builtin)
#if __has_builtin(__builtin_amdgcn_tensor_load_to_lds)
#define HAVE_TDM 1
#endif
#if __has_builtin(__builtin_amdgcn_sched_group_barrier)
#define SCHED_GROUP(m, n, id) __builtin_amdgcn_sched_group_barrier((m), (n), (id))
#endif
#endif
#ifndef SCHED_GROUP
#define SCHED_GROUP(m, n, id)
#endif

// sched_group_barrier masks
#define SG_WMMA    0x008   // MFMA/WMMA
#define SG_VMEM_RD 0x020   // VMEM read
#define SG_DS_RD   0x100   // DS read

// LDS tile: 128 rows x 64 bf16, padded to 72 bf16 (144B) pitch by TDM padding
// (pad_interval = 32 dwords = one 128B row, pad_amount = 4 dwords = 16B).
// 144B pitch keeps ds_load_b128 16B-aligned; r*36 banks mod 64 distinct for
// all 16 fragment rows -> conflict-free.
#define TILE_PITCH 72
#define TILE_ELEMS (128 * TILE_PITCH)

__device__ __forceinline__ v16bf ld16(const bf16* p0, const bf16* p1) {
    v8bf lo = *(const v8bf*)p0;
    v8bf hi = *(const v8bf*)p1;
    return __builtin_shufflevector(lo, hi, 0, 1, 2, 3, 4, 5, 6, 7,
                                           8, 9, 10, 11, 12, 13, 14, 15);
}

__device__ __forceinline__ v8f wmma_bf16(v16bf a, v16bf b, v8f c) {
    return __builtin_amdgcn_wmma_f32_16x16x32_bf16(false, a, false, b,
                                                   (short)0, c, false, false);
}

// ---------------------------------------------------------------------------
// TDM: async-load a 128x64 bf16 tile (row stride = srcStride elements) from
// global into LDS at byte offset ldsoff, padded to a 144B LDS row pitch.
// Wave-level op (EXEC ignored); tracked by TENSORcnt.
// ---------------------------------------------------------------------------
__device__ __forceinline__ void tdm_load_tile(const bf16* gsrc, unsigned ldsoff,
                                              int srcStride) {
#if HAVE_TDM
    unsigned long long ga = (unsigned long long)(uintptr_t)gsrc;
    u32x4 g0;
    g0[0] = 1u;                                                // count=1, user
    g0[1] = ldsoff;                                            // lds_addr (B)
    g0[2] = (unsigned)ga;                                      // gaddr[31:0]
    g0[3] = (unsigned)((ga >> 32) & 0x01FFFFFFu) | (2u << 30); // [56:32]|type=2
    i32x8 g1;
    // data_size=1 (2B) | pad_enable | pad_interval=4 (32 dw) | pad_amount=3 (4 dw)
    g1[0] = (1 << 16) | (1 << 20) | (4 << 22) | (3 << 25);
    g1[1] = (srcStride & 0xFFFF) << 16;                        // tensor_dim0 lo
    g1[2] = ((srcStride >> 16) & 0xFFFF);                      // dim0 hi | dim1 lo
    g1[3] = 0x10 | (64 << 16);                                 // dim1 hi | tile_dim0=64
    g1[4] = 128;                                               // tile_dim1=128
    g1[5] = srcStride;                                         // dim0_stride lo
    g1[6] = 0;
    g1[7] = 0;
    i32x4 z4 = {0, 0, 0, 0};
#if __has_include(<hip/amd_detail/amd_gfx1250_TDM.h>)
    i32x8 z8 = {0, 0, 0, 0, 0, 0, 0, 0};
    __builtin_amdgcn_tensor_load_to_lds(g0, g1, z4, z4, z8, 0);
#else
    __builtin_amdgcn_tensor_load_to_lds(g0, g1, z4, z4, 0);
#endif
#else
    (void)gsrc; (void)ldsoff; (void)srcStride;
#endif
}

#if !HAVE_TDM
// Fallback: cooperative synchronous copy into the same padded layout.
__device__ __forceinline__ void copy_tile(bf16* dst, const bf16* src, int srcStride) {
    const int row = threadIdx.x >> 1;       // 128 rows over 256 threads
    const int hf  = threadIdx.x & 1;        // 32 elements each
    v8bf* d = (v8bf*)(dst + row * TILE_PITCH + hf * 32);
    const v8bf* s = (const v8bf*)(src + (size_t)row * srcStride + hf * 32);
    d[0] = s[0]; d[1] = s[1]; d[2] = s[2]; d[3] = s[3];
}
#endif

// ---------------------------------------------------------------------------
// fp32 -> bf16 cast (vectorized x4)
// ---------------------------------------------------------------------------
__global__ void cast_f32_bf16(const float* __restrict__ in,
                              bf16* __restrict__ out, int n) {
    int i = (blockIdx.x * blockDim.x + threadIdx.x) * 4;
    if (i + 3 < n) {
        float4 v = *(const float4*)(in + i);
        out[i + 0] = (bf16)v.x;
        out[i + 1] = (bf16)v.y;
        out[i + 2] = (bf16)v.z;
        out[i + 3] = (bf16)v.w;
    }
}

// ---------------------------------------------------------------------------
// One 64-wide K step: per 32-sub-step, 18 ds_load_b128 (2 A + 16 B) pipelined
// ahead of the 8-WMMA chain.  sched_group_barrier enforces the pipeline so
// the waitcnt inserter emits partial dscnt waits.
// ---------------------------------------------------------------------------
__device__ __forceinline__ void gemm_kstep(const bf16* As, const bf16* Bs,
                                           v8f acc[8], int wid, int hl, int lm) {
#pragma unroll
    for (int s = 0; s < 2; ++s) {
        const bf16* ar = As + (wid * 16 + lm) * TILE_PITCH + 32 * s + 8 * hl;
        v16bf a = ld16(ar, ar + 16);
        v16bf bfrag[8];
#pragma unroll
        for (int i = 0; i < 8; ++i) {
            const bf16* br = Bs + (i * 16 + lm) * TILE_PITCH + 32 * s + 8 * hl;
            bfrag[i] = ld16(br, br + 16);
        }
#pragma unroll
        for (int i = 0; i < 8; ++i)
            acc[i] = wmma_bf16(a, bfrag[i], acc[i]);
    }
    // pipeline: [18 DS reads][8 WMMA] x 2
    SCHED_GROUP(SG_DS_RD, 18, 0);
    SCHED_GROUP(SG_WMMA, 8, 0);
    SCHED_GROUP(SG_DS_RD, 18, 0);
    SCHED_GROUP(SG_WMMA, 8, 0);
}

// ---------------------------------------------------------------------------
// GEMM: Y = X @ W^T.  X: [M][K] bf16 row-major, W: [N][K] bf16 row-major.
// Block = 256 threads = 8 waves; block tile 128(M) x 128(N); K-step 64.
// A/B tiles double-buffered in LDS, filled by TDM (wave0: A, wave1: B).
// mode 0: outF[m*N + n]                     (fp32, final projection)
// mode 1: outB[((n>>7)*M + m)*128 + n&127]  (head-major Q/K)
// mode 2: outB[n*M + m]                     (transposed V: Vt[h*128+d][t])
// ---------------------------------------------------------------------------
__global__ void __launch_bounds__(256, 4)
gemm_bf16_wmma(const bf16* __restrict__ X,
               const bf16* __restrict__ W,
               float* __restrict__ outF,
               bf16* __restrict__ outB,
               int M, int N, int K, int mode) {
    __shared__ bf16 smem[2][2][TILE_ELEMS];   // [buf][A/B][row*72 + k]

    const int lane = threadIdx.x & 31;
    const int wid  = threadIdx.x >> 5;
    const int hl   = lane >> 4;
    const int lm   = lane & 15;

    const int m0 = blockIdx.y * 128;
    const int n0 = blockIdx.x * 128;

    const bf16* Atile = X + (size_t)m0 * K;
    const bf16* Btile = W + (size_t)n0 * K;

    // LDS byte offsets of the 4 tiles (low 32 bits of flat addr = LDS offset)
    unsigned offA[2], offB[2];
    offA[0] = (unsigned)(uintptr_t)(void*)&smem[0][0][0];
    offB[0] = (unsigned)(uintptr_t)(void*)&smem[0][1][0];
    offA[1] = (unsigned)(uintptr_t)(void*)&smem[1][0][0];
    offB[1] = (unsigned)(uintptr_t)(void*)&smem[1][1][0];

    v8f acc[8] = {};

    // prologue: fill buffer 0
#if HAVE_TDM
    if (wid == 0) tdm_load_tile(Atile, offA[0], K);
    if (wid == 1) tdm_load_tile(Btile, offB[0], K);
#else
    copy_tile(&smem[0][0][0], Atile, K);
    copy_tile(&smem[0][1][0], Btile, K);
#endif

    int p = 0;
    for (int kb = 0; kb + 64 < K; kb += 64, p ^= 1) {
        __syncthreads();   // everyone done reading buf p^1 from previous iter
#if HAVE_TDM
        if (wid == 0) tdm_load_tile(Atile + kb + 64, offA[p ^ 1], K);
        if (wid == 1) tdm_load_tile(Btile + kb + 64, offB[p ^ 1], K);
        if (wid < 2) __builtin_amdgcn_s_wait_tensorcnt(1);  // oldest (buf p) done
#else
        copy_tile(&smem[p ^ 1][0][0], Atile + kb + 64, K);
        copy_tile(&smem[p ^ 1][1][0], Btile + kb + 64, K);
#endif
        __syncthreads();   // buf p visible to all waves
        gemm_kstep(&smem[p][0][0], &smem[p][1][0], acc, wid, hl, lm);
    }

    // final K block
    __syncthreads();
#if HAVE_TDM
    if (wid < 2) __builtin_amdgcn_s_wait_tensorcnt(0);
#endif
    __syncthreads();
    gemm_kstep(&smem[p][0][0], &smem[p][1][0], acc, wid, hl, lm);

    // C layout: lanes 0-15 -> rows r, lanes 16-31 -> rows 8+r; col = lm.
    const int mw = m0 + wid * 16;
#pragma unroll
    for (int i = 0; i < 8; ++i) {
        const int n = n0 + i * 16 + lm;
#pragma unroll
        for (int r = 0; r < 8; ++r) {
            const int m = mw + 8 * hl + r;
            const float v = acc[i][r];
            if (mode == 0) {
                outF[(size_t)m * N + n] = v;
            } else if (mode == 1) {
                outB[(((size_t)(n >> 7)) * M + m) * 128 + (n & 127)] = (bf16)v;
            } else {
                outB[(size_t)n * M + m] = (bf16)v;
            }
        }
    }
}

// ---------------------------------------------------------------------------
// RoPE on Q and K, in place.  Layout: [H][T][128].
// ---------------------------------------------------------------------------
__global__ void rope_qk(bf16* __restrict__ Q, bf16* __restrict__ K) {
    int idx = blockIdx.x * blockDim.x + threadIdx.x;   // H*T*64
    int j = idx & 63;
    int t = (idx >> 6) & (T_SEQ - 1);
    size_t row = (size_t)(idx >> 6);                   // h*T + t

    float c = 1.0f, s = 0.0f;
    if (j < 32) {
        float f = __expf(-(float)j * (9.210340371976184f / 31.0f)); // ln(1e4)
        float th = (float)t * f;
        __sincosf(th, &s, &c);
    }
    size_t p1 = row * 128 + j, p2 = p1 + 64;

    float q1 = (float)Q[p1], q2 = (float)Q[p2];
    Q[p1] = (bf16)(q1 * c + q2 * s);
    Q[p2] = (bf16)(q2 * c - q1 * s);

    float k1 = (float)K[p1], k2 = (float)K[p2];
    K[p1] = (bf16)(k1 * c + k2 * s);
    K[p2] = (bf16)(k2 * c - k1 * s);
}

// ---------------------------------------------------------------------------
// Causal flash attention.  One wave handles 16 query rows of one head.
// Q/K: [H][T][128] bf16 (post-RoPE), Vt: [H][128][T] bf16.
// Out: [T][2048] bf16.  grid = (H, T/128), block = 256 (8 waves).
// Pipeline per key block: [16 K loads][8 QK WMMA][16 V loads]...[8 PV WMMA];
// V loads are hoisted above the softmax so their latency hides under it.
// __shfl_xor lowers to ds_swizzle (DS class) so only VMEM/WMMA groups pinned.
// ---------------------------------------------------------------------------
__global__ void __launch_bounds__(256, 4)
attn_flash(const bf16* __restrict__ Q,
           const bf16* __restrict__ Kc,
           const bf16* __restrict__ Vt,
           bf16* __restrict__ O) {
    __shared__ bf16 plds[8][16][40];

    const int lane = threadIdx.x & 31;
    const int wid  = threadIdx.x >> 5;
    const int hl   = lane >> 4;
    const int lm   = lane & 15;

    const int h  = blockIdx.x;
    const int t0 = (blockIdx.y * 8 + wid) * 16;

    const bf16* Qh = Q  + (size_t)h * T_SEQ * D_HEAD;
    const bf16* Kh = Kc + (size_t)h * T_SEQ * D_HEAD;
    const bf16* Vh = Vt + (size_t)h * D_HEAD * T_SEQ;

    const bf16* qrow = Qh + (size_t)(t0 + lm) * D_HEAD;
    v16bf qf[4];
#pragma unroll
    for (int kt = 0; kt < 4; ++kt)
        qf[kt] = ld16(qrow + 32 * kt + 8 * hl, qrow + 32 * kt + 16 + 8 * hl);

    float mi[8], li[8];
#pragma unroll
    for (int r = 0; r < 8; ++r) { mi[r] = -1e30f; li[r] = 0.0f; }
    v8f oacc[8] = {};

    const float scale = 0.08838834764831845f;   // 1/sqrt(128)

    for (int kb = 0; kb <= t0 + 15; kb += 32) {
        v8f s0 = {}, s1 = {};
        const bf16* kr0 = Kh + (size_t)(kb + lm) * D_HEAD;
        const bf16* kr1 = Kh + (size_t)(kb + 16 + lm) * D_HEAD;

        // batch-preload all 8 K fragments
        v16bf kf0[4], kf1[4];
#pragma unroll
        for (int kt = 0; kt < 4; ++kt) {
            kf0[kt] = ld16(kr0 + 32 * kt + 8 * hl, kr0 + 32 * kt + 16 + 8 * hl);
            kf1[kt] = ld16(kr1 + 32 * kt + 8 * hl, kr1 + 32 * kt + 16 + 8 * hl);
        }
#pragma unroll
        for (int kt = 0; kt < 4; ++kt) {
            s0 = wmma_bf16(qf[kt], kf0[kt], s0);
            s1 = wmma_bf16(qf[kt], kf1[kt], s1);
        }

        // V fragments issued now; latency hidden under the softmax below
        v16bf vf[8];
#pragma unroll
        for (int c = 0; c < 8; ++c) {
            const bf16* vrow = Vh + (size_t)(16 * c + lm) * T_SEQ + kb;
            vf[c] = ld16(vrow + 8 * hl, vrow + 16 + 8 * hl);
        }

        const bool needmask = (kb + 31 > t0);
        const int c0 = kb + lm, c1 = kb + 16 + lm;
        float al[8];
#pragma unroll
        for (int r = 0; r < 8; ++r) {
            const int rowg = t0 + 8 * hl + r;
            float v0 = s0[r] * scale;
            float v1 = s1[r] * scale;
            if (needmask) {
                if (c0 > rowg) v0 = -1e30f;
                if (c1 > rowg) v1 = -1e30f;
            }
            float rm = fmaxf(v0, v1);
            rm = fmaxf(rm, __shfl_xor(rm, 1));
            rm = fmaxf(rm, __shfl_xor(rm, 2));
            rm = fmaxf(rm, __shfl_xor(rm, 4));
            rm = fmaxf(rm, __shfl_xor(rm, 8));
            const float mn = fmaxf(mi[r], rm);
            al[r] = __expf(mi[r] - mn);
            mi[r] = mn;
            const float p0 = __expf(v0 - mn);
            const float p1 = __expf(v1 - mn);
            float rs = p0 + p1;
            rs += __shfl_xor(rs, 1);
            rs += __shfl_xor(rs, 2);
            rs += __shfl_xor(rs, 4);
            rs += __shfl_xor(rs, 8);
            li[r] = li[r] * al[r] + rs;
            plds[wid][8 * hl + r][lm]      = (bf16)p0;
            plds[wid][8 * hl + r][16 + lm] = (bf16)p1;
        }

#pragma unroll
        for (int c = 0; c < 8; ++c)
#pragma unroll
            for (int r = 0; r < 8; ++r)
                oacc[c][r] *= al[r];

        v16bf pf = ld16(&plds[wid][lm][8 * hl], &plds[wid][lm][16 + 8 * hl]);

#pragma unroll
        for (int c = 0; c < 8; ++c)
            oacc[c] = wmma_bf16(pf, vf[c], oacc[c]);

        // pipeline: [16 K loads][8 QK WMMA][16 V loads] ... [8 PV WMMA]
        SCHED_GROUP(SG_VMEM_RD, 16, 0);
        SCHED_GROUP(SG_WMMA, 8, 0);
        SCHED_GROUP(SG_VMEM_RD, 16, 0);
        SCHED_GROUP(SG_WMMA, 8, 0);
    }

    float invl[8];
#pragma unroll
    for (int r = 0; r < 8; ++r) invl[r] = 1.0f / li[r];
#pragma unroll
    for (int c = 0; c < 8; ++c) {
        const int colg = h * D_HEAD + 16 * c + lm;
#pragma unroll
        for (int r = 0; r < 8; ++r) {
            const int rowg = t0 + 8 * hl + r;
            O[(size_t)rowg * D_MODEL + colg] = (bf16)(oacc[c][r] * invl[r]);
        }
    }
}

// ---------------------------------------------------------------------------
// Launcher
// ---------------------------------------------------------------------------
extern "C" void kernel_launch(void* const* d_in, const int* in_sizes, int n_in,
                              void* d_out, int out_size, void* d_ws, size_t ws_size,
                              hipStream_t stream) {
    const float* x  = (const float*)d_in[0];
    const float* Wq = (const float*)d_in[1];
    const float* Wk = (const float*)d_in[2];
    const float* Wv = (const float*)d_in[3];
    const float* Wo = (const float*)d_in[4];

    const int TD = T_SEQ * D_MODEL;
    const int DD = D_MODEL * D_MODEL;

    bf16* ws  = (bf16*)d_ws;
    bf16* xb  = ws;            // [T][D]
    bf16* wqb = xb  + TD;      // [D][D]
    bf16* wkb = wqb + DD;
    bf16* wvb = wkb + DD;
    bf16* wob = wvb + DD;
    bf16* qb  = wob + DD;      // [H][T][128]
    bf16* kb  = qb  + TD;      // [H][T][128]
    bf16* vt  = kb  + TD;      // [H][128][T]
    bf16* ob  = vt  + TD;      // [T][D]

    cast_f32_bf16<<<TD / (256 * 4), 256, 0, stream>>>(x,  xb,  TD);
    cast_f32_bf16<<<DD / (256 * 4), 256, 0, stream>>>(Wq, wqb, DD);
    cast_f32_bf16<<<DD / (256 * 4), 256, 0, stream>>>(Wk, wkb, DD);
    cast_f32_bf16<<<DD / (256 * 4), 256, 0, stream>>>(Wv, wvb, DD);
    cast_f32_bf16<<<DD / (256 * 4), 256, 0, stream>>>(Wo, wob, DD);

    dim3 gg(D_MODEL / 128, T_SEQ / 128);
    gemm_bf16_wmma<<<gg, 256, 0, stream>>>(xb, wqb, nullptr, qb,
                                           T_SEQ, D_MODEL, D_MODEL, 1);
    gemm_bf16_wmma<<<gg, 256, 0, stream>>>(xb, wkb, nullptr, kb,
                                           T_SEQ, D_MODEL, D_MODEL, 1);
    gemm_bf16_wmma<<<gg, 256, 0, stream>>>(xb, wvb, nullptr, vt,
                                           T_SEQ, D_MODEL, D_MODEL, 2);

    rope_qk<<<(N_HEADS * T_SEQ * 64) / 256, 256, 0, stream>>>(qb, kb);

    attn_flash<<<dim3(N_HEADS, T_SEQ / 128), 256, 0, stream>>>(qb, kb, vt, ob);

    gemm_bf16_wmma<<<gg, 256, 0, stream>>>(ob, wob, (float*)d_out, nullptr,
                                           T_SEQ, D_MODEL, D_MODEL, 0);
}
